// LinearAttention_54193897341234
// MI455X (gfx1250) — matrix operand order repaired and checked by reading, not compiled, not run on hardware
//
#include <hip/hip_runtime.h>
#include <hip/hip_bf16.h>
#include <math.h>

// ---------- types ----------
typedef __bf16         bf16x16 __attribute__((ext_vector_type(16)));
typedef float          f32x8   __attribute__((ext_vector_type(8)));
typedef float          f32x4   __attribute__((ext_vector_type(4)));
typedef unsigned int   u32x4   __attribute__((ext_vector_type(4)));
typedef unsigned short u16x4   __attribute__((ext_vector_type(4)));

union Frag { u32x4 u4[2]; bf16x16 v; };

#define WMMA_BF16(a, b, c) \
  __builtin_amdgcn_wmma_f32_16x16x32_bf16(false, (a), false, (b), (short)0, (c), false, false)

__device__ __forceinline__ unsigned short f2bf(float f) {
  unsigned int u = __float_as_uint(f);
  return (unsigned short)((u + 0x7fffu + ((u >> 16) & 1u)) >> 16);  // RNE
}
__device__ __forceinline__ float bf2f(unsigned short h) {
  return __uint_as_float(((unsigned int)h) << 16);
}

// gfx1250 async global->LDS copy (16B per lane), ASYNCcnt-tracked.
__device__ __forceinline__ void async_ld_b128(unsigned lds_byte_addr, const void* gptr) {
  asm volatile("global_load_async_to_lds_b128 %0, %1, off"
               :: "v"(lds_byte_addr), "v"((unsigned long long)(size_t)gptr)
               : "memory");
}
__device__ __forceinline__ void wait_async_le6() {
  asm volatile("s_wait_asynccnt 0x6" ::: "memory");
}

// ---------- problem constants ----------
#define BATCH 4
#define SEQ   4096
#define EMB   1024
#define HEADS 16
#define HDIM  64
#define MTOT  (BATCH * SEQ)  // 16384

// GEMM tiling
#define BM   256
#define BN   128
#define BK   32
#define LDW  40                      // padded row stride (halfwords): 80B, 16B-aligned
#define STAGE_HW (BM * LDW + BN * LDW)  // halfwords per stage buffer

// ---------- kernel 1: fp32 -> bf16 cast (linear) ----------
__global__ __launch_bounds__(256) void cast_bf16_kernel(const float* __restrict__ in,
                                                        unsigned short* __restrict__ out,
                                                        int n) {
  int i4 = (blockIdx.x * 256 + threadIdx.x) * 4;
  if (i4 >= n) return;
  f32x4 f = *(const f32x4*)(in + i4);
  u16x4 o;
  o.x = f2bf(f.x); o.y = f2bf(f.y); o.z = f2bf(f.z); o.w = f2bf(f.w);
  *(u16x4*)(out + i4) = o;
}

// ---------- kernel 1b: fp32 (R x C) -> bf16 transposed (C x R) ----------
__global__ __launch_bounds__(256) void transpose_cast_kernel(const float* __restrict__ in,
                                                             unsigned short* __restrict__ out,
                                                             int R, int Ccols) {
  __shared__ unsigned short t[32][33];
  const int tid = threadIdx.x;
  const int tx = tid & 31, ty = tid >> 5;  // 32 x 8
  const int c0 = blockIdx.x * 32, r0 = blockIdx.y * 32;
#pragma unroll
  for (int i = 0; i < 32; i += 8)
    t[tx][ty + i] = f2bf(in[(size_t)(r0 + ty + i) * Ccols + c0 + tx]);
  __syncthreads();
#pragma unroll
  for (int i = 0; i < 32; i += 8)
    out[(size_t)(c0 + ty + i) * R + r0 + tx] = t[ty + i][tx];
}

// ---------- kernel 2: qkv = x @ W_qkv (Wt pre-transposed), elu+1 on q/k, head scatter ----------
// A: x_bf (M x K row-major). Bt: W^T (3072 x 1024 row-major bf16).
__global__ __launch_bounds__(256) void qkv_gemm_kernel(const unsigned short* __restrict__ A,
                                                       const unsigned short* __restrict__ Bt,
                                                       unsigned short* __restrict__ q_t,  // (B,H,N,D)
                                                       unsigned short* __restrict__ k_t,  // (B,H,D,N)
                                                       unsigned short* __restrict__ v_t)  // (B,H,D,N)
{
  constexpr int K = EMB;
  __shared__ __align__(16) unsigned short lds[2 * STAGE_HW];

  const int tid = threadIdx.x;
  const int wid = tid >> 5, lane = tid & 31;
  const int ln15 = lane & 15, half = lane >> 4;
  const int m0 = blockIdx.y * BM, n0 = blockIdx.x * BN;
  const int wm = (wid >> 1) * 64, wn = (wid & 1) * 64;
  const unsigned ldsBase = (unsigned)(size_t)&lds[0];

  auto stage = [&](int kb, int st) {
    const unsigned base = ldsBase + (unsigned)(st * STAGE_HW * 2);
#pragma unroll
    for (int i = 0; i < 4; ++i) {  // A tile: 256x32, 16B chunks
      int idx = i * 256 + tid;
      int row = idx >> 2, c8 = (idx & 3) * 8;
      async_ld_b128(base + (unsigned)(row * LDW + c8) * 2,
                    A + (size_t)(m0 + row) * K + kb + c8);
    }
#pragma unroll
    for (int i = 0; i < 2; ++i) {  // B tile: 128 rows (n) x 32 (k)
      int idx = i * 256 + tid;
      int n = idx >> 2, c8 = (idx & 3) * 8;
      async_ld_b128(base + (unsigned)((BM * LDW) + n * LDW + c8) * 2,
                    Bt + (size_t)(n0 + n) * K + kb + c8);
    }
  };

  const f32x8 zero8 = {0.f, 0.f, 0.f, 0.f, 0.f, 0.f, 0.f, 0.f};
  f32x8 acc[4][4];
#pragma unroll
  for (int i = 0; i < 4; ++i)
#pragma unroll
    for (int j = 0; j < 4; ++j) acc[i][j] = zero8;

  stage(0, 0);
  for (int it = 0; it < K / BK; ++it) {
    stage(((it + 1) * BK) & (K - 1), (it + 1) & 1);  // prefetch next (wraps harmlessly)
    wait_async_le6();                                // previous stage landed
    __syncthreads();

    const unsigned short* Ab = lds + (it & 1) * STAGE_HW;
    const unsigned short* Bb = Ab + BM * LDW;
    Frag af[4], bf[4];
#pragma unroll
    for (int mt = 0; mt < 4; ++mt) {
      const unsigned short* rp = Ab + (wm + mt * 16 + ln15) * LDW;
      af[mt].u4[0] = *(const u32x4*)(rp + half * 8);
      af[mt].u4[1] = *(const u32x4*)(rp + 16 + half * 8);
    }
#pragma unroll
    for (int nt = 0; nt < 4; ++nt) {
      const unsigned short* cp = Bb + (wn + nt * 16 + ln15) * LDW;
      bf[nt].u4[0] = *(const u32x4*)(cp + half * 8);
      bf[nt].u4[1] = *(const u32x4*)(cp + 16 + half * 8);
    }
#pragma unroll
    for (int mt = 0; mt < 4; ++mt)
#pragma unroll
      for (int nt = 0; nt < 4; ++nt)
        acc[mt][nt] = WMMA_BF16(af[mt].v, bf[nt].v, acc[mt][nt]);
    __syncthreads();
  }

  // epilogue: q/k/v split, elu+1 on q,k, scatter to head layouts
#pragma unroll
  for (int mt = 0; mt < 4; ++mt) {
#pragma unroll
    for (int nt = 0; nt < 4; ++nt) {
      int ng = n0 + wn + nt * 16 + ln15;
      int sec = ng >> 10;  // 0:q 1:k 2:v (uniform per block)
      int c = ng & 1023;
      int h = c >> 6, dcol = c & 63;
#pragma unroll
      for (int r = 0; r < 8; ++r) {
        int mg = m0 + wm + mt * 16 + r + 8 * half;
        int bb = mg >> 12, nr = mg & 4095;
        float val = acc[mt][nt][r];
        if (sec == 0) {
          float e = val > 0.f ? val + 1.f : __expf(val);
          q_t[(((size_t)(bb * HEADS + h)) * SEQ + nr) * HDIM + dcol] = f2bf(e);
        } else if (sec == 1) {
          float e = val > 0.f ? val + 1.f : __expf(val);
          k_t[(((size_t)(bb * HEADS + h)) * HDIM + dcol) * SEQ + nr] = f2bf(e);
        } else {
          v_t[(((size_t)(bb * HEADS + h)) * HDIM + dcol) * SEQ + nr] = f2bf(val);
        }
      }
    }
  }
}

// ---------- kernel 3: kv = k^T v (64x64 per head) + ksum ----------
__global__ __launch_bounds__(256) void kv_ksum_kernel(const unsigned short* __restrict__ k_t,
                                                      const unsigned short* __restrict__ v_t,
                                                      unsigned short* __restrict__ kvT,
                                                      float* __restrict__ ksum) {
  const int bh = blockIdx.x;
  const unsigned short* kp = k_t + (size_t)bh * HDIM * SEQ;
  const unsigned short* vp = v_t + (size_t)bh * HDIM * SEQ;
  __shared__ float kvS[HDIM * HDIM];
  __shared__ float ksS[256];

  const int tid = threadIdx.x, wid = tid >> 5, lane = tid & 31;
  const int ln15 = lane & 15, half = lane >> 4;

  for (int i = tid; i < HDIM * HDIM; i += 256) kvS[i] = 0.f;

  {  // ksum: thread handles quarter of one d-row
    int d = tid >> 2, qtr = tid & 3;
    const unsigned short* rp = kp + (size_t)d * SEQ + qtr * 1024;
    float s = 0.f;
    for (int i = 0; i < 1024; i += 8) {
      u32x4 u = *(const u32x4*)(rp + i);
      const unsigned short* us = (const unsigned short*)&u;
#pragma unroll
      for (int j = 0; j < 8; ++j) s += bf2f(us[j]);
    }
    ksS[tid] = s;
  }
  __syncthreads();
  if (tid < HDIM)
    ksum[bh * HDIM + tid] = ksS[tid * 4] + ksS[tid * 4 + 1] + ksS[tid * 4 + 2] + ksS[tid * 4 + 3];

  const f32x8 zero8 = {0.f, 0.f, 0.f, 0.f, 0.f, 0.f, 0.f, 0.f};
  f32x8 acc[4][4];
#pragma unroll
  for (int i = 0; i < 4; ++i)
#pragma unroll
    for (int j = 0; j < 4; ++j) acc[i][j] = zero8;

  const int nbase = wid * (SEQ / 8);
  for (int kk = 0; kk < SEQ / 8; kk += 32) {
    int nco = nbase + kk;
    Frag af[4], bf[4];
#pragma unroll
    for (int mt = 0; mt < 4; ++mt) {
      const unsigned short* rp = kp + (size_t)(mt * 16 + ln15) * SEQ + nco;
      __builtin_prefetch(rp + 32, 0, 1);  // next K-strip -> global_prefetch_b8
      af[mt].u4[0] = *(const u32x4*)(rp + half * 8);
      af[mt].u4[1] = *(const u32x4*)(rp + 16 + half * 8);
    }
#pragma unroll
    for (int nt = 0; nt < 4; ++nt) {
      const unsigned short* cp = vp + (size_t)(nt * 16 + ln15) * SEQ + nco;
      __builtin_prefetch(cp + 32, 0, 1);
      bf[nt].u4[0] = *(const u32x4*)(cp + half * 8);
      bf[nt].u4[1] = *(const u32x4*)(cp + 16 + half * 8);
    }
#pragma unroll
    for (int mt = 0; mt < 4; ++mt)
#pragma unroll
      for (int nt = 0; nt < 4; ++nt)
        acc[mt][nt] = WMMA_BF16(af[mt].v, bf[nt].v, acc[mt][nt]);
  }

#pragma unroll
  for (int mt = 0; mt < 4; ++mt)
#pragma unroll
    for (int nt = 0; nt < 4; ++nt)
#pragma unroll
      for (int r = 0; r < 8; ++r) {
        int dd = mt * 16 + r + 8 * half, ee = nt * 16 + ln15;
        atomicAdd(&kvS[dd * HDIM + ee], acc[mt][nt][r]);
      }
  __syncthreads();

  for (int i = tid; i < HDIM * HDIM; i += 256) {  // write kv transposed: kvT[e][d]
    int e = i >> 6, dd = i & 63;
    kvT[(size_t)bh * HDIM * HDIM + i] = f2bf(kvS[dd * HDIM + e]);
  }
}

// ---------- kernel 4: num = q @ kv, normalize, write attn (B,N,C) bf16 ----------
__global__ __launch_bounds__(256) void num_norm_kernel(const unsigned short* __restrict__ q_t,
                                                       const unsigned short* __restrict__ kvT,
                                                       const float* __restrict__ ksum,
                                                       unsigned short* __restrict__ attn) {
  const int bh = blockIdx.y;
  const int bb = bh >> 4, h = bh & 15;
  const int n0 = blockIdx.x * 256;
  const unsigned short* qp = q_t + (size_t)bh * SEQ * HDIM;
  const unsigned short* kvp = kvT + (size_t)bh * HDIM * HDIM;
  __shared__ float ksS[HDIM];
  __shared__ float normS[256];

  const int tid = threadIdx.x, wid = tid >> 5, lane = tid & 31;
  const int ln15 = lane & 15, half = lane >> 4;
  if (tid < HDIM) ksS[tid] = ksum[bh * HDIM + tid];
  __syncthreads();

  {  // normalizer: one row per thread
    const unsigned short* rp = qp + (size_t)(n0 + tid) * HDIM;
    float s = 0.f;
#pragma unroll
    for (int i = 0; i < HDIM; i += 8) {
      u32x4 u = *(const u32x4*)(rp + i);
      const unsigned short* us = (const unsigned short*)&u;
#pragma unroll
      for (int j = 0; j < 8; ++j) s += bf2f(us[j]) * ksS[i + j];
    }
    normS[tid] = 1.0f / (s + 1e-6f);
  }
  __syncthreads();

  const f32x8 zero8 = {0.f, 0.f, 0.f, 0.f, 0.f, 0.f, 0.f, 0.f};
  f32x8 acc[2][4];
#pragma unroll
  for (int i = 0; i < 2; ++i)
#pragma unroll
    for (int j = 0; j < 4; ++j) acc[i][j] = zero8;

  const int wm = wid * 32;
#pragma unroll
  for (int kk = 0; kk < HDIM; kk += 32) {
    Frag af[2], bf[4];
#pragma unroll
    for (int mt = 0; mt < 2; ++mt) {
      const unsigned short* rp = qp + (size_t)(n0 + wm + mt * 16 + ln15) * HDIM + kk;
      af[mt].u4[0] = *(const u32x4*)(rp + half * 8);
      af[mt].u4[1] = *(const u32x4*)(rp + 16 + half * 8);
    }
#pragma unroll
    for (int nt = 0; nt < 4; ++nt) {
      const unsigned short* cp = kvp + (size_t)(nt * 16 + ln15) * HDIM + kk;
      bf[nt].u4[0] = *(const u32x4*)(cp + half * 8);
      bf[nt].u4[1] = *(const u32x4*)(cp + 16 + half * 8);
    }
#pragma unroll
    for (int mt = 0; mt < 2; ++mt)
#pragma unroll
      for (int nt = 0; nt < 4; ++nt)
        acc[mt][nt] = WMMA_BF16(af[mt].v, bf[nt].v, acc[mt][nt]);
  }

#pragma unroll
  for (int mt = 0; mt < 2; ++mt)
#pragma unroll
    for (int nt = 0; nt < 4; ++nt)
#pragma unroll
      for (int r = 0; r < 8; ++r) {
        int rloc = wm + mt * 16 + r + 8 * half;
        int ng = n0 + rloc;
        int ee = nt * 16 + ln15;
        float val = acc[mt][nt][r] * normS[rloc];
        attn[(((size_t)bb * SEQ + ng) * HEADS + h) * HDIM + ee] = f2bf(val);
      }
}

// ---------- kernel 5: out = attn @ W_out + b_out (Wt pre-transposed, f32 result) ----------
__global__ __launch_bounds__(256) void out_gemm_kernel(const unsigned short* __restrict__ A,
                                                       const unsigned short* __restrict__ Bt,
                                                       const float* __restrict__ bias,
                                                       float* __restrict__ out) {
  constexpr int K = EMB, NCOLS = EMB;
  __shared__ __align__(16) unsigned short lds[2 * STAGE_HW];

  const int tid = threadIdx.x;
  const int wid = tid >> 5, lane = tid & 31;
  const int ln15 = lane & 15, half = lane >> 4;
  const int m0 = blockIdx.y * BM, n0 = blockIdx.x * BN;
  const int wm = (wid >> 1) * 64, wn = (wid & 1) * 64;
  const unsigned ldsBase = (unsigned)(size_t)&lds[0];

  auto stage = [&](int kb, int st) {
    const unsigned base = ldsBase + (unsigned)(st * STAGE_HW * 2);
#pragma unroll
    for (int i = 0; i < 4; ++i) {
      int idx = i * 256 + tid;
      int row = idx >> 2, c8 = (idx & 3) * 8;
      async_ld_b128(base + (unsigned)(row * LDW + c8) * 2,
                    A + (size_t)(m0 + row) * K + kb + c8);
    }
#pragma unroll
    for (int i = 0; i < 2; ++i) {
      int idx = i * 256 + tid;
      int n = idx >> 2, c8 = (idx & 3) * 8;
      async_ld_b128(base + (unsigned)((BM * LDW) + n * LDW + c8) * 2,
                    Bt + (size_t)(n0 + n) * K + kb + c8);
    }
  };

  const f32x8 zero8 = {0.f, 0.f, 0.f, 0.f, 0.f, 0.f, 0.f, 0.f};
  f32x8 acc[4][4];
#pragma unroll
  for (int i = 0; i < 4; ++i)
#pragma unroll
    for (int j = 0; j < 4; ++j) acc[i][j] = zero8;

  stage(0, 0);
  for (int it = 0; it < K / BK; ++it) {
    stage(((it + 1) * BK) & (K - 1), (it + 1) & 1);
    wait_async_le6();
    __syncthreads();

    const unsigned short* Ab = lds + (it & 1) * STAGE_HW;
    const unsigned short* Bb = Ab + BM * LDW;
    Frag af[4], bf[4];
#pragma unroll
    for (int mt = 0; mt < 4; ++mt) {
      const unsigned short* rp = Ab + (wm + mt * 16 + ln15) * LDW;
      af[mt].u4[0] = *(const u32x4*)(rp + half * 8);
      af[mt].u4[1] = *(const u32x4*)(rp + 16 + half * 8);
    }
#pragma unroll
    for (int nt = 0; nt < 4; ++nt) {
      const unsigned short* cp = Bb + (wn + nt * 16 + ln15) * LDW;
      bf[nt].u4[0] = *(const u32x4*)(cp + half * 8);
      bf[nt].u4[1] = *(const u32x4*)(cp + 16 + half * 8);
    }
#pragma unroll
    for (int mt = 0; mt < 4; ++mt)
#pragma unroll
      for (int nt = 0; nt < 4; ++nt)
        acc[mt][nt] = WMMA_BF16(af[mt].v, bf[nt].v, acc[mt][nt]);
    __syncthreads();
  }

#pragma unroll
  for (int mt = 0; mt < 4; ++mt)
#pragma unroll
    for (int nt = 0; nt < 4; ++nt) {
      int ng = n0 + wn + nt * 16 + ln15;
      float bv = bias[ng];
#pragma unroll
      for (int r = 0; r < 8; ++r) {
        int mg = m0 + wm + mt * 16 + r + 8 * half;
        out[(size_t)mg * NCOLS + ng] = acc[mt][nt][r] + bv;
      }
    }
}

// ---------- host launcher ----------
extern "C" void kernel_launch(void* const* d_in, const int* in_sizes, int n_in,
                              void* d_out, int out_size, void* d_ws, size_t ws_size,
                              hipStream_t stream) {
  const float* x    = (const float*)d_in[0];
  const float* Wqkv = (const float*)d_in[1];
  const float* Wout = (const float*)d_in[2];
  const float* bout = (const float*)d_in[3];

  constexpr size_t SZ_X    = (size_t)MTOT * EMB;
  constexpr size_t SZ_WQKV = (size_t)EMB * 3 * EMB;
  constexpr size_t SZ_WOUT = (size_t)EMB * EMB;
  constexpr size_t SZ_HT   = (size_t)BATCH * HEADS * SEQ * HDIM;
  constexpr size_t SZ_KV   = (size_t)BATCH * HEADS * HDIM * HDIM;

  unsigned short* x_bf    = (unsigned short*)d_ws;
  unsigned short* wqkv_bt = x_bf + SZ_X;     // W_qkv^T: (3072 x 1024) bf16
  unsigned short* wout_bt = wqkv_bt + SZ_WQKV;  // W_out^T: (1024 x 1024) bf16
  unsigned short* q_t     = wout_bt + SZ_WOUT;
  unsigned short* k_t     = q_t + SZ_HT;
  unsigned short* v_t     = k_t + SZ_HT;
  unsigned short* kvT     = v_t + SZ_HT;
  float*          ksum    = (float*)(kvT + SZ_KV);
  unsigned short* attn    = x_bf;  // reuse: x_bf dead after qkv_gemm

  // 1) casts (+ weight transposes, done once)
  cast_bf16_kernel<<<(int)(SZ_X / 4 / 256), 256, 0, stream>>>(x, x_bf, (int)SZ_X);
  transpose_cast_kernel<<<dim3(3 * EMB / 32, EMB / 32), 256, 0, stream>>>(Wqkv, wqkv_bt, EMB, 3 * EMB);
  transpose_cast_kernel<<<dim3(EMB / 32, EMB / 32), 256, 0, stream>>>(Wout, wout_bt, EMB, EMB);

  // 2) qkv projection + activation + head scatter
  qkv_gemm_kernel<<<dim3(3 * EMB / BN, MTOT / BM), 256, 0, stream>>>(x_bf, wqkv_bt, q_t, k_t, v_t);

  // 3) per-head kv (D x D) and ksum
  kv_ksum_kernel<<<BATCH * HEADS, 256, 0, stream>>>(k_t, v_t, kvT, ksum);

  // 4) numerator + normalization -> attn (B,N,C) bf16
  num_norm_kernel<<<dim3(SEQ / 256, BATCH * HEADS), 256, 0, stream>>>(q_t, kvT, ksum, attn);

  // 5) output projection
  out_gemm_kernel<<<dim3(EMB / BN, MTOT / BM), 256, 0, stream>>>(attn, wout_bt, bout, (float*)d_out);
}
